// Sector_11416023072802
// MI455X (gfx1250) — compile-verified, matrix operand order checked by reading, not run on hardware
//
#include <hip/hip_runtime.h>

// ---------------------------------------------------------------------------
// Types
// ---------------------------------------------------------------------------
typedef unsigned short u16;
typedef __attribute__((ext_vector_type(16))) __bf16 v16bf;
typedef __attribute__((ext_vector_type(8)))  float  v8f;
typedef __attribute__((ext_vector_type(4)))  float  f32x4;

union TileU { v16bf v; f32x4 f4[2]; };

#define S_LEN 256
#define B_SZ  32
#define E_DIM 768
#define H_DIM 1024
#define G_DIM 4096   // 4*H
#define Q_DIM 64
#define T_DIM 11
#define N_PCA 16
#define SB    8192   // S*B

#define RNN_BLOCKS 128   // persistent kernel: 128 blocks x 256 thr = 1024 waves

// ---------------------------------------------------------------------------
// bf16 helpers (bit-level, no scalar __bf16 arithmetic)
// ---------------------------------------------------------------------------
__device__ inline u16 f2bf(float f) {
    union { float f; unsigned u; } x; x.f = f;
    unsigned r = x.u + 0x7FFFu + ((x.u >> 16) & 1u);   // RNE
    return (u16)(r >> 16);
}
__device__ inline float bf2f(u16 h) {
    union { unsigned u; float f; } x; x.u = ((unsigned)h) << 16;
    return x.f;
}

__device__ inline float waveRed(float v) {
#pragma unroll
    for (int o = 16; o > 0; o >>= 1) v += __shfl_xor(v, o, 32);
    return v;
}

// ---------------------------------------------------------------------------
// WMMA tile loader: 16-bit A (16x32, row r = lane&15) or B (32x16, col = lane&15)
// stored row-major [r][K]. Lane loads K ranges [h*8,h*8+8) and [16+h*8,...+8),
// h = lane>>4, as two 16-byte loads -> matches CDNA5 16-bit A/B VGPR packing.
// ---------------------------------------------------------------------------
__device__ inline v16bf load_tile(const u16* base, int ld, int r, int k0, int lane) {
    const u16* p = base + (size_t)r * ld + k0 + ((lane >> 4) << 3);
    TileU u;
    u.f4[0] = *(const f32x4*)(p);
    u.f4[1] = *(const f32x4*)(p + 16);
    return u.v;
}

__device__ inline v8f wmma_bf16(v16bf a, v16bf b, v8f c) {
    return __builtin_amdgcn_wmma_f32_16x16x32_bf16(false, a, false, b, (short)0, c,
                                                   false, false);
}

// ---------------------------------------------------------------------------
// Device-wide generation barrier (persistent LSTM kernel)
// ---------------------------------------------------------------------------
__device__ inline void grid_sync(unsigned* cnt, unsigned* gen, unsigned nb) {
    __syncthreads();
    if (threadIdx.x == 0) {
        unsigned g = __hip_atomic_load(gen, __ATOMIC_ACQUIRE, __HIP_MEMORY_SCOPE_AGENT);
        unsigned my = __hip_atomic_fetch_add(cnt, 1u, __ATOMIC_ACQ_REL, __HIP_MEMORY_SCOPE_AGENT);
        if (my == nb - 1u) {
            __hip_atomic_store(cnt, 0u, __ATOMIC_RELAXED, __HIP_MEMORY_SCOPE_AGENT);
            __hip_atomic_fetch_add(gen, 1u, __ATOMIC_ACQ_REL, __HIP_MEMORY_SCOPE_AGENT);
        } else {
            while (__hip_atomic_load(gen, __ATOMIC_ACQUIRE, __HIP_MEMORY_SCOPE_AGENT) == g) {
                __builtin_amdgcn_s_sleep(2);
            }
        }
    }
    __syncthreads();
}

// ---------------------------------------------------------------------------
// f32 -> bf16 convert
// ---------------------------------------------------------------------------
__global__ void cvt_bf16_kernel(const float* __restrict__ src, u16* __restrict__ dst, int n) {
    int i = blockIdx.x * blockDim.x + threadIdx.x;
    if (i < n) dst[i] = f2bf(src[i]);
}

// ---------------------------------------------------------------------------
// LayerNorm over rows -> bf16 output (128 threads per row)
// ---------------------------------------------------------------------------
__global__ void ln_bf16_kernel(const float* __restrict__ x, const float* __restrict__ g,
                               const float* __restrict__ be, u16* __restrict__ y, int dim) {
    int row = blockIdx.x;
    const float* xr = x + (size_t)row * dim;
    int tid = threadIdx.x;
    __shared__ float red[4];

    float s = 0.f;
    for (int i = tid; i < dim; i += 128) s += xr[i];
    s = waveRed(s);
    if ((tid & 31) == 0) red[tid >> 5] = s;
    __syncthreads();
    float mu = (red[0] + red[1] + red[2] + red[3]) / (float)dim;
    __syncthreads();

    float v = 0.f;
    for (int i = tid; i < dim; i += 128) { float d = xr[i] - mu; v += d * d; }
    v = waveRed(v);
    if ((tid & 31) == 0) red[tid >> 5] = v;
    __syncthreads();
    float rinv = rsqrtf((red[0] + red[1] + red[2] + red[3]) / (float)dim + 1e-5f);

    for (int i = tid; i < dim; i += 128)
        y[(size_t)row * dim + i] = f2bf(g[i] * (xr[i] - mu) * rinv + be[i]);
}

// ---------------------------------------------------------------------------
// xg = ln(x)[rev?] @ Wih^T + (bih+bhh)   -> bf16 out [M][N]
// A: bf16 [M][K], W: bf16 [N][K].
// 2x2 register blocking: each wave computes a 32x32 output block
// (2 A-frags + 2 B-frags -> 4 WMMAs per k-step, halving L2 fragment traffic).
// ---------------------------------------------------------------------------
__global__ void gemm_xg_kernel(const u16* __restrict__ A, const u16* __restrict__ W,
                               const float* __restrict__ b1, const float* __restrict__ b2,
                               u16* __restrict__ C, int M, int N, int K, int rev) {
    int wave = (blockIdx.x * blockDim.x + threadIdx.x) >> 5;
    int lane = threadIdx.x & 31;
    int ntile = N >> 5;                  // 32-wide supertiles
    int tm = wave / ntile, tn = wave % ntile;
    if (tm * 32 >= M) return;
    int m0 = tm * 32, n0 = tn * 32;

    int ar0 = m0 + (lane & 15);
    int ar1 = m0 + 16 + (lane & 15);
    if (rev) {
        int s0 = ar0 >> 5, bb0 = ar0 & 31; ar0 = ((S_LEN - 1 - s0) << 5) | bb0;
        int s1 = ar1 >> 5, bb1 = ar1 & 31; ar1 = ((S_LEN - 1 - s1) << 5) | bb1;
    }
    int br0 = n0 + (lane & 15);
    int br1 = n0 + 16 + (lane & 15);

    v8f acc00 = {0.f, 0.f, 0.f, 0.f, 0.f, 0.f, 0.f, 0.f};
    v8f acc01 = {0.f, 0.f, 0.f, 0.f, 0.f, 0.f, 0.f, 0.f};
    v8f acc10 = {0.f, 0.f, 0.f, 0.f, 0.f, 0.f, 0.f, 0.f};
    v8f acc11 = {0.f, 0.f, 0.f, 0.f, 0.f, 0.f, 0.f, 0.f};
    for (int k = 0; k < K; k += 32) {
        v16bf a0 = load_tile(A, K, ar0, k, lane);
        v16bf a1 = load_tile(A, K, ar1, k, lane);
        v16bf w0 = load_tile(W, K, br0, k, lane);
        v16bf w1 = load_tile(W, K, br1, k, lane);
        acc00 = wmma_bf16(a0, w0, acc00);
        acc01 = wmma_bf16(a0, w1, acc01);
        acc10 = wmma_bf16(a1, w0, acc10);
        acc11 = wmma_bf16(a1, w1, acc11);
    }
    int half = lane >> 4;
    int colL = lane & 15;
#pragma unroll
    for (int j = 0; j < 2; j++) {
        int col = n0 + j * 16 + colL;
        float bias = b1[col] + b2[col];
#pragma unroll
        for (int i = 0; i < 2; i++) {
            v8f acc = (i == 0) ? ((j == 0) ? acc00 : acc01)
                               : ((j == 0) ? acc10 : acc11);
#pragma unroll
            for (int r = 0; r < 8; r++) {
                int row = m0 + i * 16 + r + half * 8;
                C[(size_t)row * N + col] = f2bf(acc[r] + bias);
            }
        }
    }
}

// ---------------------------------------------------------------------------
// Persistent bidirectional LSTM:
//   for t in 0..255:
//     phase 1 (1024 waves): gates[dir][b][g] = h[dir] @ Whh[dir]^T + xg[dir][t]
//       wave -> (dir, tm in 0..1, tn in 0..255), one 16x16 WMMA tile, K=1024
//     grid barrier
//     phase 2 (all 32768 threads): LSTM cell update, write h (bf16+f32), c
//     grid barrier
// Whh (8.4 MB/dir, bf16) streams from the 192MB L2 once per step; hstate
// (64 KB/dir) stays hot in cache. Single launch kills 512-node graph overhead.
// ---------------------------------------------------------------------------
__global__ void lstm_seq_kernel(u16* __restrict__ hstate, float* __restrict__ cstate,
                                const u16* __restrict__ WhhF, const u16* __restrict__ WhhB,
                                const u16* __restrict__ xgF, const u16* __restrict__ xgB,
                                float* __restrict__ gates, float* __restrict__ hsF,
                                float* __restrict__ hsB, unsigned* barCnt, unsigned* barGen) {
    int tid = threadIdx.x;
    int gidx = blockIdx.x * blockDim.x + tid;
    int nthreads = gridDim.x * blockDim.x;
    int wave = gidx >> 5;
    int lane = tid & 31;

    int dir = wave >> 9;
    int rem = wave & 511;
    int tm = rem >> 8;        // 0..1
    int tn = rem & 255;       // 0..255
    const u16* A  = hstate + (size_t)dir * B_SZ * H_DIM;
    const u16* W  = (dir == 0) ? WhhF : WhhB;
    const u16* xg = (dir == 0) ? xgF : xgB;

    int m0 = tm * 16, n0 = tn * 16;
    int arow = m0 + (lane & 15);
    int brow = n0 + (lane & 15);
    int col = n0 + (lane & 15);
    int half = lane >> 4;
    float* go = gates + (size_t)dir * B_SZ * G_DIM;

    for (int t = 0; t < S_LEN; t++) {
        // ---- phase 1: recurrent GEMM tile ----
        v8f acc = {0.f, 0.f, 0.f, 0.f, 0.f, 0.f, 0.f, 0.f};
        for (int k = 0; k < H_DIM; k += 32) {
            __builtin_prefetch((const void*)(W + (size_t)brow * H_DIM + k + 64), 0, 1);
            v16bf a = load_tile(A, H_DIM, arow, k, lane);
            v16bf b = load_tile(W, H_DIM, brow, k, lane);
            acc = wmma_bf16(a, b, acc);
        }
        const u16* xr = xg + (size_t)t * B_SZ * G_DIM;
#pragma unroll
        for (int r = 0; r < 8; r++) {
            int row = m0 + r + half * 8;
            go[(size_t)row * G_DIM + col] = acc[r] + bf2f(xr[(size_t)row * G_DIM + col]);
        }

        grid_sync(barCnt, barGen, gridDim.x);

        // ---- phase 2: cell update (2*32*1024 elements) ----
        for (int e = gidx; e < 2 * B_SZ * H_DIM; e += nthreads) {
            int d2 = e >> 15;
            int r2 = e & 32767;
            int b2 = r2 >> 10;
            int h2 = r2 & 1023;
            const float* gr = gates + (size_t)d2 * B_SZ * G_DIM + (size_t)b2 * G_DIM;
            float ig = gr[h2], fg = gr[H_DIM + h2], gg = gr[2 * H_DIM + h2], og = gr[3 * H_DIM + h2];
            float si = 1.f / (1.f + expf(-ig));
            float sf = 1.f / (1.f + expf(-fg));
            float so = 1.f / (1.f + expf(-og));
            float c = cstate[e];
            c = sf * c + si * tanhf(gg);
            float hv = so * tanhf(c);
            cstate[e] = c;
            hstate[e] = f2bf(hv);
            int sout = (d2 == 0) ? t : (S_LEN - 1 - t);
            float* hs = (d2 == 0) ? hsF : hsB;
            hs[((size_t)sout * B_SZ + b2) * H_DIM + h2] = hv;
        }

        grid_sync(barCnt, barGen, gridDim.x);
    }
}

// ---------------------------------------------------------------------------
// Projection GEMM: xcat[:, cofs:cofs+64] = tanh(ln(hs) @ Wsq^T + bsq)
// M=8192, N=64, K=1024, ldc=128
// ---------------------------------------------------------------------------
__global__ void gemm_proj_kernel(const u16* __restrict__ A, const u16* __restrict__ Wsq,
                                 const float* __restrict__ bsq, float* __restrict__ xcat,
                                 int cofs) {
    int wave = (blockIdx.x * blockDim.x + threadIdx.x) >> 5;
    int lane = threadIdx.x & 31;
    int tm = wave >> 2;        // 512 tiles
    int tn = wave & 3;         // 4 tiles
    int m0 = tm * 16, n0 = tn * 16;
    int arow = m0 + (lane & 15);
    int brow = n0 + (lane & 15);
    v8f acc = {0.f, 0.f, 0.f, 0.f, 0.f, 0.f, 0.f, 0.f};
    for (int k = 0; k < H_DIM; k += 32) {
        v16bf a = load_tile(A, H_DIM, arow, k, lane);
        v16bf b = load_tile(Wsq, H_DIM, brow, k, lane);
        acc = wmma_bf16(a, b, acc);
    }
    int col = n0 + (lane & 15);
    float bias = bsq[col];
    int half = lane >> 4;
#pragma unroll
    for (int r = 0; r < 8; r++) {
        int row = m0 + r + half * 8;
        xcat[(size_t)row * 128 + cofs + col] = tanhf(acc[r] + bias);
    }
}

// ---------------------------------------------------------------------------
// Topic head: out[sb][t] = ln(xcat)[sb] . Wtop[t] + btop[t]
// ---------------------------------------------------------------------------
__global__ void topic_kernel(const u16* __restrict__ xcln, const float* __restrict__ Wtop,
                             const float* __restrict__ btop, float* __restrict__ out) {
    int idx = blockIdx.x * blockDim.x + threadIdx.x;
    if (idx >= SB * T_DIM) return;
    int sb = idx / T_DIM, t = idx % T_DIM;
    const u16* xr = xcln + (size_t)sb * 128;
    const float* w = Wtop + (size_t)t * 128;
    float s = btop[t];
#pragma unroll 8
    for (int i = 0; i < 128; i++) s += bf2f(xr[i]) * w[i];
    out[idx] = s;
}

// ---------------------------------------------------------------------------
// Per-batch mean + covariance (64x64) of projection outputs (pre-LN)
// blockIdx.x = dir*32 + b ; 256 threads
// ---------------------------------------------------------------------------
__global__ void pca_cov_kernel(const float* __restrict__ xcat, float* __restrict__ meanbuf,
                               float* __restrict__ cov) {
    __shared__ float psum[4][64];
    __shared__ float meanS[64];
    __shared__ u16 cS[S_LEN * 64];      // centered values, bf16 bits (32 KB)
    int mat = blockIdx.x;
    int dir = mat >> 5, b = mat & 31;
    int tid = threadIdx.x;
    int f = tid & 63, grp = tid >> 6;

    float s = 0.f;
    for (int ss = grp * 64; ss < grp * 64 + 64; ss++)
        s += xcat[((size_t)ss * B_SZ + b) * 128 + dir * 64 + f];
    psum[grp][f] = s;
    __syncthreads();
    if (tid < 64) meanS[tid] = (psum[0][tid] + psum[1][tid] + psum[2][tid] + psum[3][tid]) / (float)S_LEN;
    __syncthreads();

    for (int idx = tid; idx < S_LEN * 64; idx += 256) {
        int ss = idx >> 6, ff = idx & 63;
        cS[idx] = f2bf(xcat[((size_t)ss * B_SZ + b) * 128 + dir * 64 + ff] - meanS[ff]);
    }
    __syncthreads();

    for (int pair = tid; pair < 64 * 64; pair += 256) {
        int i = pair >> 6, j = pair & 63;
        float acc = 0.f;
        for (int ss = 0; ss < S_LEN; ss++)
            acc += bf2f(cS[ss * 64 + i]) * bf2f(cS[ss * 64 + j]);
        cov[(size_t)mat * 4096 + pair] = acc / (float)S_LEN;
    }
    if (tid < 64) meanbuf[(size_t)mat * 64 + tid] = meanS[tid];
}

// ---------------------------------------------------------------------------
// Cyclic Jacobi eigensolver for 64x64 symmetric, 64 threads per matrix.
// Writes top-16 eigenvectors (descending eigenvalue): topv[mat][f][n]
// ---------------------------------------------------------------------------
__global__ void jacobi_kernel(const float* __restrict__ cov, float* __restrict__ topv) {
    __shared__ float Am[64][64];
    __shared__ float Vm[64][64];
    __shared__ int idxS[16];
    int tid = threadIdx.x;
    int mat = blockIdx.x;
    const float* src = cov + (size_t)mat * 4096;
    for (int i = 0; i < 64; i++) {
        Am[i][tid] = src[i * 64 + tid];
        Vm[i][tid] = (i == tid) ? 1.f : 0.f;
    }
    __syncthreads();

    for (int sweep = 0; sweep < 8; sweep++) {
        for (int p = 0; p < 63; p++) {
            for (int q = p + 1; q < 64; q++) {
                float apq = Am[p][q];
                float app = Am[p][p], aqq = Am[q][q];
                float c = 1.f, s = 0.f;
                if (fabsf(apq) > 1e-12f) {
                    float tau = (aqq - app) / (2.f * apq);
                    float tt = ((tau >= 0.f) ? 1.f : -1.f) / (fabsf(tau) + sqrtf(1.f + tau * tau));
                    c = rsqrtf(1.f + tt * tt);
                    s = tt * c;
                }
                __syncthreads();
                float aip = Am[tid][p], aiq = Am[tid][q];
                Am[tid][p] = c * aip - s * aiq;
                Am[tid][q] = s * aip + c * aiq;
                float vip = Vm[tid][p], viq = Vm[tid][q];
                Vm[tid][p] = c * vip - s * viq;
                Vm[tid][q] = s * vip + c * viq;
                __syncthreads();
                float apj = Am[p][tid], aqj = Am[q][tid];
                Am[p][tid] = c * apj - s * aqj;
                Am[q][tid] = s * apj + c * aqj;
                __syncthreads();
            }
        }
    }

    if (tid == 0) {
        unsigned long long used = 0ULL;
        for (int n = 0; n < N_PCA; n++) {
            int best = 0; float bv = -1e30f;
            for (int j = 0; j < 64; j++)
                if (!((used >> j) & 1ULL) && Am[j][j] > bv) { bv = Am[j][j]; best = j; }
            used |= (1ULL << best);
            idxS[n] = best;
        }
    }
    __syncthreads();
    for (int n = 0; n < N_PCA; n++)
        topv[(size_t)mat * 64 * N_PCA + tid * N_PCA + n] = Vm[tid][idxS[n]];
}

// ---------------------------------------------------------------------------
// PCA projection: eproj[mat][s][n] = (x - mean) . top[:, n]
// ---------------------------------------------------------------------------
__global__ void pca_proj_kernel(const float* __restrict__ xcat, const float* __restrict__ meanbuf,
                                const float* __restrict__ topv, float* __restrict__ eproj) {
    int idx = blockIdx.x * blockDim.x + threadIdx.x;
    if (idx >= 64 * S_LEN * N_PCA) return;
    int n = idx & 15;
    int s = (idx >> 4) & 255;
    int mat = idx >> 12;
    int dir = mat >> 5, b = mat & 31;
    const float* mean = meanbuf + (size_t)mat * 64;
    const float* V = topv + (size_t)mat * 64 * N_PCA;
    float acc = 0.f;
    for (int f = 0; f < 64; f++)
        acc += (xcat[((size_t)s * B_SZ + b) * 128 + dir * 64 + f] - mean[f]) * V[f * N_PCA + n];
    eproj[idx] = acc;
}

// ---------------------------------------------------------------------------
// 11x11 Gaussian smoothing (same padding) over (s, n)
// ---------------------------------------------------------------------------
__global__ void gauss_kernel(const float* __restrict__ e, const float* __restrict__ kern,
                             float* __restrict__ out) {
    int idx = blockIdx.x * blockDim.x + threadIdx.x;
    if (idx >= 64 * S_LEN * N_PCA) return;
    int n = idx & 15;
    int s = (idx >> 4) & 255;
    int mat = idx >> 12;
    float acc = 0.f;
    for (int u = 0; u < 11; u++) {
        int ss = s + u - 5;
        if (ss < 0 || ss >= S_LEN) continue;
        for (int v = 0; v < 11; v++) {
            int nn = n + v - 5;
            if (nn < 0 || nn >= N_PCA) continue;
            acc += kern[u * 11 + v] * e[((size_t)mat * S_LEN + ss) * N_PCA + nn];
        }
    }
    out[idx] = acc;
}

// ---------------------------------------------------------------------------
// d[b][s] = sqrt(adjcos_f * adjcos_b), s in [0,255)
// ---------------------------------------------------------------------------
__global__ void adjd_kernel(const float* __restrict__ esm, float* __restrict__ dbuf) {
    int idx = blockIdx.x * blockDim.x + threadIdx.x;
    if (idx >= B_SZ * (S_LEN - 1)) return;
    int b = idx / (S_LEN - 1), s = idx % (S_LEN - 1);
    float prod = 1.f;
    for (int dir = 0; dir < 2; dir++) {
        const float* p = esm + (((size_t)(dir * B_SZ + b) * S_LEN) + s) * N_PCA;
        const float* q = p + N_PCA;
        float num = 0.f, na = 0.f, nb = 0.f;
#pragma unroll
        for (int i = 0; i < N_PCA; i++) { num += p[i] * q[i]; na += p[i] * p[i]; nb += q[i] * q[i]; }
        prod *= num / (sqrtf(na) * sqrtf(nb));
    }
    dbuf[idx] = sqrtf(prod);
}

// ---------------------------------------------------------------------------
// argmax over s per batch -> out (as float)
// ---------------------------------------------------------------------------
__global__ void argmax_kernel(const float* __restrict__ dbuf, float* __restrict__ out) {
    __shared__ float sv[256];
    __shared__ int si[256];
    int b = blockIdx.x;
    int tid = threadIdx.x;
    float bv = -1e30f; int bi = 0;
    for (int s = tid; s < S_LEN - 1; s += 256) {
        float v = dbuf[b * (S_LEN - 1) + s];
        if (v > bv) { bv = v; bi = s; }
    }
    sv[tid] = bv; si[tid] = bi;
    __syncthreads();
    for (int o = 128; o > 0; o >>= 1) {
        if (tid < o) {
            if (sv[tid + o] > sv[tid] || (sv[tid + o] == sv[tid] && si[tid + o] < si[tid])) {
                sv[tid] = sv[tid + o]; si[tid] = si[tid + o];
            }
        }
        __syncthreads();
    }
    if (tid == 0) out[b] = (float)si[0];
}

// ---------------------------------------------------------------------------
// Host launcher
// ---------------------------------------------------------------------------
extern "C" void kernel_launch(void* const* d_in, const int* in_sizes, int n_in,
                              void* d_out, int out_size, void* d_ws, size_t ws_size,
                              hipStream_t stream) {
    const float* x_in  = (const float*)d_in[0];
    const float* WihF  = (const float*)d_in[1];
    const float* WhhF  = (const float*)d_in[2];
    const float* bihF  = (const float*)d_in[3];
    const float* bhhF  = (const float*)d_in[4];
    const float* WihB  = (const float*)d_in[5];
    const float* WhhB  = (const float*)d_in[6];
    const float* bihB  = (const float*)d_in[7];
    const float* bhhB  = (const float*)d_in[8];
    const float* g_in  = (const float*)d_in[9];
    const float* be_in = (const float*)d_in[10];
    const float* g_fw  = (const float*)d_in[11];
    const float* be_fw = (const float*)d_in[12];
    const float* g_bk  = (const float*)d_in[13];
    const float* be_bk = (const float*)d_in[14];
    const float* g_sq  = (const float*)d_in[15];
    const float* be_sq = (const float*)d_in[16];
    const float* Wsq   = (const float*)d_in[17];
    const float* bsq   = (const float*)d_in[18];
    const float* Wtop  = (const float*)d_in[19];
    const float* btop  = (const float*)d_in[20];
    const float* gk    = (const float*)d_in[21];
    float* out = (float*)d_out;

    // workspace carve-up (256B aligned)
    char* cur = (char*)d_ws;
    auto alloc = [&](size_t bytes) -> void* {
        void* r = (void*)cur;
        cur += (bytes + 255) & ~(size_t)255;
        return r;
    };
    u16* xln    = (u16*)alloc((size_t)SB * E_DIM * 2);
    u16* WihFb  = (u16*)alloc((size_t)G_DIM * E_DIM * 2);
    u16* WihBb  = (u16*)alloc((size_t)G_DIM * E_DIM * 2);
    u16* WhhFb  = (u16*)alloc((size_t)G_DIM * H_DIM * 2);
    u16* WhhBb  = (u16*)alloc((size_t)G_DIM * H_DIM * 2);
    u16* Wsqb   = (u16*)alloc((size_t)Q_DIM * H_DIM * 2);
    u16* xgF    = (u16*)alloc((size_t)SB * G_DIM * 2);
    u16* xgB    = (u16*)alloc((size_t)SB * G_DIM * 2);
    float* hsF  = (float*)alloc((size_t)SB * H_DIM * 4);
    float* hsB  = (float*)alloc((size_t)SB * H_DIM * 4);
    u16* hlnF   = (u16*)alloc((size_t)SB * H_DIM * 2);
    u16* hlnB   = (u16*)alloc((size_t)SB * H_DIM * 2);
    u16* hstate = (u16*)alloc((size_t)2 * B_SZ * H_DIM * 2);
    float* cstate = (float*)alloc((size_t)2 * B_SZ * H_DIM * 4);
    float* gates  = (float*)alloc((size_t)2 * B_SZ * G_DIM * 4);
    float* xcat   = (float*)alloc((size_t)SB * 128 * 4);
    u16* xcln     = (u16*)alloc((size_t)SB * 128 * 2);
    float* meanb  = (float*)alloc((size_t)64 * 64 * 4);
    float* cov    = (float*)alloc((size_t)64 * 4096 * 4);
    float* topv   = (float*)alloc((size_t)64 * 64 * N_PCA * 4);
    float* eproj  = (float*)alloc((size_t)64 * S_LEN * N_PCA * 4);
    float* esm    = (float*)alloc((size_t)64 * S_LEN * N_PCA * 4);
    float* dbuf   = (float*)alloc((size_t)B_SZ * (S_LEN - 1) * 4);
    unsigned* barrier = (unsigned*)alloc(2 * sizeof(unsigned));

    // 1) convert weights to bf16
    {
        int n1 = G_DIM * E_DIM, n2 = G_DIM * H_DIM, n3 = Q_DIM * H_DIM;
        cvt_bf16_kernel<<<(n1 + 255) / 256, 256, 0, stream>>>(WihF, WihFb, n1);
        cvt_bf16_kernel<<<(n1 + 255) / 256, 256, 0, stream>>>(WihB, WihBb, n1);
        cvt_bf16_kernel<<<(n2 + 255) / 256, 256, 0, stream>>>(WhhF, WhhFb, n2);
        cvt_bf16_kernel<<<(n2 + 255) / 256, 256, 0, stream>>>(WhhB, WhhBb, n2);
        cvt_bf16_kernel<<<(n3 + 255) / 256, 256, 0, stream>>>(Wsq, Wsqb, n3);
    }

    // 2) input layernorm -> bf16
    ln_bf16_kernel<<<SB, 128, 0, stream>>>(x_in, g_in, be_in, xln, E_DIM);

    // 3) xg GEMMs (WMMA, 2x2 register-blocked): M=8192, N=4096, K=768
    {
        int waves = (SB / 32) * (G_DIM / 32);
        int blocks = waves * 32 / 256;
        gemm_xg_kernel<<<blocks, 256, 0, stream>>>(xln, WihFb, bihF, bhhF, xgF,
                                                   SB, G_DIM, E_DIM, 0);
        gemm_xg_kernel<<<blocks, 256, 0, stream>>>(xln, WihBb, bihB, bhhB, xgB,
                                                   SB, G_DIM, E_DIM, 1);
    }

    // 4) LSTM recurrence: single persistent kernel, on-device sequencing
    hipMemsetAsync(hstate, 0, (size_t)2 * B_SZ * H_DIM * 2, stream);
    hipMemsetAsync(cstate, 0, (size_t)2 * B_SZ * H_DIM * 4, stream);
    hipMemsetAsync(barrier, 0, 2 * sizeof(unsigned), stream);
    lstm_seq_kernel<<<RNN_BLOCKS, 256, 0, stream>>>(hstate, cstate, WhhFb, WhhBb,
                                                    xgF, xgB, gates, hsF, hsB,
                                                    barrier, barrier + 1);

    // 5) hidden layernorms -> bf16
    ln_bf16_kernel<<<SB, 128, 0, stream>>>(hsF, g_fw, be_fw, hlnF, H_DIM);
    ln_bf16_kernel<<<SB, 128, 0, stream>>>(hsB, g_bk, be_bk, hlnB, H_DIM);

    // 6) projection GEMMs (WMMA, tanh epilogue) -> xcat f32 [8192][128]
    {
        int waves = (SB / 16) * (Q_DIM / 16);
        int blocks = waves * 32 / 256;
        gemm_proj_kernel<<<blocks, 256, 0, stream>>>(hlnF, Wsqb, bsq, xcat, 0);
        gemm_proj_kernel<<<blocks, 256, 0, stream>>>(hlnB, Wsqb, bsq, xcat, Q_DIM);
    }

    // 7) concat layernorm -> bf16, then topic head
    ln_bf16_kernel<<<SB, 128, 0, stream>>>(xcat, g_sq, be_sq, xcln, 128);
    topic_kernel<<<(SB * T_DIM + 255) / 256, 256, 0, stream>>>(xcln, Wtop, btop, out);

    // 8) PCA per (dir, batch): mean+cov, Jacobi eigh, project
    pca_cov_kernel<<<64, 256, 0, stream>>>(xcat, meanb, cov);
    jacobi_kernel<<<64, 64, 0, stream>>>(cov, topv);
    pca_proj_kernel<<<(64 * S_LEN * N_PCA + 255) / 256, 256, 0, stream>>>(xcat, meanb, topv, eproj);

    // 9) Gaussian smoothing, adjacent cosine, argmax
    gauss_kernel<<<(64 * S_LEN * N_PCA + 255) / 256, 256, 0, stream>>>(eproj, gk, esm);
    adjd_kernel<<<(B_SZ * (S_LEN - 1) + 255) / 256, 256, 0, stream>>>(esm, dbuf);
    argmax_kernel<<<B_SZ, 256, 0, stream>>>(dbuf, out + SB * T_DIM);

    (void)in_sizes; (void)n_in; (void)out_size; (void)ws_size;
}
